// MultiHeadsAttention_56831007261404
// MI455X (gfx1250) — matrix operand order repaired and checked by reading, not compiled, not run on hardware
//
#include <hip/hip_runtime.h>
#include <hip/hip_bf16.h>

// ---------------------------------------------------------------------------
// MHA for gfx1250 (MI455X): all GEMM work on v_wmma_f32_16x16x32_f16 (wave32),
// global->LDS staging via GLOBAL_LOAD_ASYNC_TO_LDS_B128 (ASYNCcnt) when the
// toolchain exposes the builtin, double-buffered.
// ---------------------------------------------------------------------------

typedef __attribute__((ext_vector_type(16))) _Float16 v16h;
typedef __attribute__((ext_vector_type(8)))  float    v8f;
typedef __attribute__((ext_vector_type(4)))  _Float16 v4h;
typedef __attribute__((ext_vector_type(4)))  int      v4i;

#define D_MODEL 2048
#define HEADS   16
#define D_K     128
#define BATCH   4
#define SEQ     2048
#define ROWS    (BATCH * SEQ)          // 8192
#define PPAD    40                     // P-tile stride (32+8): 80B rows, 16B-aligned
#define GPAD    72                     // GEMM LDS stride (64+8): 144B rows, 16B-aligned

#ifndef __has_builtin
#define __has_builtin(x) 0
#endif
#if __has_builtin(__builtin_amdgcn_global_load_async_to_lds_b128) && \
    __has_builtin(__builtin_amdgcn_s_wait_asynccnt)
#define ASYNC_LDS 1
#else
#define ASYNC_LDS 0
#endif

union U16x2 { uint4 u[2]; v16h v; };

__device__ __forceinline__ v8f vzero8() {
  v8f z;
#pragma unroll
  for (int i = 0; i < 8; ++i) z[i] = 0.0f;
  return z;
}

// A-fragment (16Mx32K, f16): lane l holds row (l&15); K elems
//   {koff..koff+7} U {koff+16..koff+23}, koff = (l>>4)*8.  Two 16B loads.
__device__ __forceinline__ v16h load_afrag(const _Float16* base, int ld) {
  const int l = threadIdx.x & 31;
  const _Float16* p = base + (size_t)(l & 15) * ld + ((l >> 4) * 8);
  U16x2 r;
  r.u[0] = *(const uint4*)(p);
  r.u[1] = *(const uint4*)(p + 16);
  return r.v;
}

// B-fragment (32Kx16N, f16) from K-major storage W[n][k]: lane l holds logical
// column n=(l&15), K rows {koff..koff+7} U {koff+16..koff+23}, koff = (l>>4)*8.
__device__ __forceinline__ v16h load_bfrag(const _Float16* base, int ld) {
  const int l = threadIdx.x & 31;
  const _Float16* p = base + (size_t)(l & 15) * ld + ((l >> 4) * 8);
  U16x2 r;
  r.u[0] = *(const uint4*)(p);
  r.u[1] = *(const uint4*)(p + 16);
  return r.v;
}

// 16B global -> LDS copy: async on CDNA5 when available.
// Builtin signature (from probe): (global int4*, shared int4*, imm offset, imm cpol).
__device__ __forceinline__ void cp16(const _Float16* g, _Float16* l) {
#if ASYNC_LDS
  __builtin_amdgcn_global_load_async_to_lds_b128(
      (__attribute__((address_space(1))) v4i*)g,
      (__attribute__((address_space(3))) v4i*)l, 0, 0);
#else
  *(uint4*)l = *(const uint4*)g;
#endif
}

// ---------------------------------------------------------------------------
// fp32 -> f16 convert, vectorized x4
// ---------------------------------------------------------------------------
__global__ void cvt_f32_f16(const float* __restrict__ in, _Float16* __restrict__ out, int n4) {
  int i = blockIdx.x * blockDim.x + threadIdx.x;
  if (i >= n4) return;
  float4 f = ((const float4*)in)[i];
  v4h h;
  h[0] = (_Float16)f.x; h[1] = (_Float16)f.y; h[2] = (_Float16)f.z; h[3] = (_Float16)f.w;
  ((v4h*)out)[i] = h;
}

// ---------------------------------------------------------------------------
// NT GEMM: C[m,n] = sum_k A[m,k] * W[n,k], M=8192, N=K=2048, f16 in, f32 acc.
// Block: 128x128 tile, 256 threads = 8 waves (2x4), wave tile 64x32.
// K-step 64, double-buffered async global->LDS staging.
// MODE 0: f16 out, [b,h,s,dk] layout (Q,K).  MODE 1: f16 out, [b,h,dk,s] (V^T).
// MODE 2: f32 out, row-major [m][n] (final projection).
// ---------------------------------------------------------------------------
template <int MODE>
__global__ __launch_bounds__(256) void gemm_nt_f16(const _Float16* __restrict__ A,
                                                   const _Float16* __restrict__ W,
                                                   float* __restrict__ Cf,
                                                   _Float16* __restrict__ Ch) {
  __shared__ _Float16 As[2][128 * GPAD];
  __shared__ _Float16 Bs[2][128 * GPAD];
  const int tid  = threadIdx.x;
  const int wave = tid >> 5;
  const int wm   = wave >> 2;   // 0..1
  const int wn   = wave & 3;    // 0..3
  const int bm   = blockIdx.y * 128;
  const int bn   = blockIdx.x * 128;

  v8f acc[4][2];
#pragma unroll
  for (int mi = 0; mi < 4; ++mi)
#pragma unroll
    for (int ni = 0; ni < 2; ++ni) acc[mi][ni] = vzero8();

  // Stage one 128x64 A tile + 128x64 B tile into buffer `buf`.
  // Per wave: 8 async-to-LDS b128 instructions (4 A + 4 B).
  auto stage = [&](int k0, int buf) {
#pragma unroll
    for (int p = 0; p < 4; ++p) {
      int ci = p * 256 + tid;            // 1024 chunks of 8 f16
      int r  = ci >> 3;                  // 0..127
      int c  = (ci & 7) * 8;             // 0..56
      const _Float16* ag = A + (size_t)(bm + r) * D_MODEL + k0 + c;
      const _Float16* bg = W + (size_t)(bn + r) * D_MODEL + k0 + c;
      cp16(ag, &As[buf][r * GPAD + c]);
      cp16(bg, &Bs[buf][r * GPAD + c]);
    }
  };

  stage(0, 0);
  for (int k0 = 0, it = 0; k0 < D_MODEL; k0 += 64, ++it) {
    const int  cur      = it & 1;
    const bool has_next = (k0 + 64) < D_MODEL;
    if (has_next) stage(k0 + 64, cur ^ 1);   // overlaps with this tile's math
#if ASYNC_LDS
    if (has_next) __builtin_amdgcn_s_wait_asynccnt(8);  // current tile done
    else          __builtin_amdgcn_s_wait_asynccnt(0);
#endif
    __syncthreads();

#pragma unroll
    for (int ks = 0; ks < 2; ++ks) {          // two 32-wide K sub-steps
      v16h af[4], bf[2];
#pragma unroll
      for (int mi = 0; mi < 4; ++mi)
        af[mi] = load_afrag(&As[cur][(wm * 64 + mi * 16) * GPAD + ks * 32], GPAD);
#pragma unroll
      for (int ni = 0; ni < 2; ++ni)
        bf[ni] = load_bfrag(&Bs[cur][(wn * 32 + ni * 16) * GPAD + ks * 32], GPAD);
#pragma unroll
      for (int mi = 0; mi < 4; ++mi)
#pragma unroll
        for (int ni = 0; ni < 2; ++ni)
          acc[mi][ni] = __builtin_amdgcn_wmma_f32_16x16x32_f16(
              false, af[mi], false, bf[ni], (short)0, acc[mi][ni], false, false);
    }
    __syncthreads();   // buffer cur may be overwritten next iteration
  }

  const int l     = tid & 31;
  const int mlane = (l >> 4) * 8;
  const int nlane = l & 15;
#pragma unroll
  for (int mi = 0; mi < 4; ++mi)
#pragma unroll
    for (int ni = 0; ni < 2; ++ni)
#pragma unroll
      for (int r = 0; r < 8; ++r) {
        int m = bm + wm * 64 + mi * 16 + r + mlane;
        int n = bn + wn * 32 + ni * 16 + nlane;
        float v = acc[mi][ni][r];
        if (MODE == 2) {
          Cf[(size_t)m * D_MODEL + n] = v;
        } else {
          int b = m >> 11, s = m & (SEQ - 1);
          int h = n >> 7, kk = n & (D_K - 1);
          if (MODE == 0)
            Ch[(((size_t)b * HEADS + h) * SEQ + s) * D_K + kk] = (_Float16)v;
          else
            Ch[(((size_t)b * HEADS + h) * D_K + kk) * SEQ + s] = (_Float16)v;
        }
      }
}

// ---------------------------------------------------------------------------
// Causal flash attention. Grid (S/128, H, B), 256 threads = 8 waves; each wave
// owns 16 q-rows. Q [b,h,s,dk], K [b,h,s,dk], V^T [b,h,dk,s] (all f16).
// Online softmax in fp32; P transposed C-layout -> A-layout via per-wave LDS.
// mh out: [b*S + q][h*D_K + v] f16.
// ---------------------------------------------------------------------------
__global__ __launch_bounds__(256) void flash_attn(const _Float16* __restrict__ Q,
                                                  const _Float16* __restrict__ Kg,
                                                  const _Float16* __restrict__ Vt,
                                                  _Float16* __restrict__ mh) {
  __shared__ _Float16 Ps[8][16 * PPAD];   // per-wave 16x32 P tile (padded)

  const int tid   = threadIdx.x;
  const int wave  = tid >> 5;
  const int l     = tid & 31;
  const int mlane = (l >> 4) * 8;
  const int nlane = l & 15;

  const int b  = blockIdx.z;
  const int h  = blockIdx.y;
  const int qw = blockIdx.x * 128 + wave * 16;   // first q-row of this wave

  const _Float16* Qb = Q  + ((size_t)b * HEADS + h) * SEQ * D_K;
  const _Float16* Kb = Kg + ((size_t)b * HEADS + h) * SEQ * D_K;
  const _Float16* Vb = Vt + ((size_t)b * HEADS + h) * D_K * SEQ;

  v16h qf[4];
#pragma unroll
  for (int kt = 0; kt < 4; ++kt)
    qf[kt] = load_afrag(Qb + (size_t)qw * D_K + kt * 32, D_K);

  v8f accO[8];
#pragma unroll
  for (int nt = 0; nt < 8; ++nt) accO[nt] = vzero8();
  float mrow[8], lrow[8];
#pragma unroll
  for (int r = 0; r < 8; ++r) { mrow[r] = -1e30f; lrow[r] = 0.0f; }

  const float qscale = 0.08838834764831845f;     // 1/sqrt(128)
  const int nsteps = (qw + 15) / 32 + 1;         // causal bound

  for (int ts = 0; ts < nsteps; ++ts) {
    const int t0 = ts * 32;
    if (ts + 1 < nsteps) {                       // warm next K/V tiles
      __builtin_prefetch(Kb + (size_t)(t0 + 32 + l) * D_K, 0, 1);
      __builtin_prefetch(Vb + (size_t)(l * 4) * SEQ + t0 + 32, 0, 1);
    }

    v8f s[2];
#pragma unroll
    for (int hf = 0; hf < 2; ++hf) {
      const int tt0 = t0 + hf * 16;
      v8f sa = vzero8();
#pragma unroll
      for (int kt = 0; kt < 4; ++kt) {
        v16h kb = load_bfrag(Kb + (size_t)tt0 * D_K + kt * 32, D_K);
        sa = __builtin_amdgcn_wmma_f32_16x16x32_f16(false, qf[kt], false, kb,
                                                    (short)0, sa, false, false);
      }
      if (tt0 + 15 > qw) {                       // diagonal tile: causal mask
#pragma unroll
        for (int r = 0; r < 8; ++r) {
          float v = sa[r] * qscale;
          if (tt0 + nlane > qw + r + mlane) v = -1e30f;
          sa[r] = v;
        }
      } else {
#pragma unroll
        for (int r = 0; r < 8; ++r) sa[r] *= qscale;
      }
      s[hf] = sa;
    }

    // Online softmax row stats (rows live across the 16-lane column groups).
    float sc[8];
#pragma unroll
    for (int r = 0; r < 8; ++r) {
      float tm = fmaxf(s[0][r], s[1][r]);
#pragma unroll
      for (int off = 1; off < 16; off <<= 1) tm = fmaxf(tm, __shfl_xor(tm, off, 32));
      float nm = fmaxf(mrow[r], tm);
      sc[r] = __expf(mrow[r] - nm);
      mrow[r] = nm;
      float p0 = __expf(s[0][r] - nm);
      float p1 = __expf(s[1][r] - nm);
      s[0][r] = p0; s[1][r] = p1;
      float rs = p0 + p1;
#pragma unroll
      for (int off = 1; off < 16; off <<= 1) rs += __shfl_xor(rs, off, 32);
      lrow[r] = lrow[r] * sc[r] + rs;
    }
#pragma unroll
    for (int nt = 0; nt < 8; ++nt)
#pragma unroll
      for (int r = 0; r < 8; ++r) accO[nt][r] *= sc[r];

    // P: C-layout -> A-layout via per-wave LDS (16x32 f16, stride PPAD).
    _Float16* ps = &Ps[wave][0];
#pragma unroll
    for (int hf = 0; hf < 2; ++hf)
#pragma unroll
      for (int r = 0; r < 8; ++r)
        ps[(r + mlane) * PPAD + nlane + hf * 16] = (_Float16)s[hf][r];
    __asm__ __volatile__("s_wait_dscnt 0" ::: "memory");
    v16h pf = load_afrag(ps, PPAD);

    // O += P(16x32) x V(32x128): 8 n-tiles; V^T rows are t-contiguous.
#pragma unroll
    for (int nt = 0; nt < 8; ++nt) {
      v16h vb = load_bfrag(Vb + (size_t)(nt * 16) * SEQ + t0, SEQ);
      accO[nt] = __builtin_amdgcn_wmma_f32_16x16x32_f16(false, pf, false, vb,
                                                        (short)0, accO[nt], false, false);
    }
  }

  float inv[8];
#pragma unroll
  for (int r = 0; r < 8; ++r) inv[r] = 1.0f / lrow[r];
#pragma unroll
  for (int nt = 0; nt < 8; ++nt)
#pragma unroll
    for (int r = 0; r < 8; ++r) {
      size_t row = (size_t)b * SEQ + qw + r + mlane;
      mh[row * D_MODEL + h * D_K + nt * 16 + nlane] = (_Float16)(accO[nt][r] * inv[r]);
    }
}

// ---------------------------------------------------------------------------
// Host-side orchestration
// ---------------------------------------------------------------------------
extern "C" void kernel_launch(void* const* d_in, const int* in_sizes, int n_in,
                              void* d_out, int out_size, void* d_ws, size_t ws_size,
                              hipStream_t stream) {
  const float* x  = (const float*)d_in[0];
  const float* WQ = (const float*)d_in[1];
  const float* WK = (const float*)d_in[2];
  const float* WV = (const float*)d_in[3];
  const float* WO = (const float*)d_in[4];

  const size_t SZ_X = (size_t)ROWS * D_MODEL;      // 16,777,216
  const size_t SZ_W = (size_t)D_MODEL * D_MODEL;   //  4,194,304

  _Float16* ws = (_Float16*)d_ws;
  _Float16* xh  = ws;                    // [8192][2048]   (reused as mh later)
  _Float16* wqh = xh  + SZ_X;
  _Float16* wkh = wqh + SZ_W;
  _Float16* wvh = wkh + SZ_W;
  _Float16* woh = wvh + SZ_W;
  _Float16* qh  = woh + SZ_W;            // [b,h,s,dk]
  _Float16* kh  = qh  + SZ_X;            // [b,h,s,dk]
  _Float16* vt  = kh  + SZ_X;            // [b,h,dk,s]
  _Float16* mhb = xh;                    // alias: x dead after QKV projections

  // 1) fp32 -> f16
  {
    int n4 = (int)(SZ_X / 4);
    cvt_f32_f16<<<(n4 + 255) / 256, 256, 0, stream>>>(x, xh, n4);
    int w4 = (int)(SZ_W / 4);
    cvt_f32_f16<<<(w4 + 255) / 256, 256, 0, stream>>>(WQ, wqh, w4);
    cvt_f32_f16<<<(w4 + 255) / 256, 256, 0, stream>>>(WK, wkh, w4);
    cvt_f32_f16<<<(w4 + 255) / 256, 256, 0, stream>>>(WV, wvh, w4);
    cvt_f32_f16<<<(w4 + 255) / 256, 256, 0, stream>>>(WO, woh, w4);
  }

  dim3 ggrid(D_MODEL / 128, ROWS / 128);   // (16, 64)
  // 2) projections (NT GEMMs)
  gemm_nt_f16<0><<<ggrid, 256, 0, stream>>>(xh, wqh, nullptr, qh);
  gemm_nt_f16<0><<<ggrid, 256, 0, stream>>>(xh, wkh, nullptr, kh);
  gemm_nt_f16<1><<<ggrid, 256, 0, stream>>>(xh, wvh, nullptr, vt);

  // 3) causal flash attention -> mh (aliases xh)
  dim3 agrid(SEQ / 128, HEADS, BATCH);     // (16, 16, 4)
  flash_attn<<<agrid, 256, 0, stream>>>(qh, kh, vt, mhb);

  // 4) output projection -> fp32 d_out
  gemm_nt_f16<2><<<ggrid, 256, 0, stream>>>(mhb, woh, (float*)d_out, nullptr);
}
